// Codebook_90804198572334
// MI455X (gfx1250) — matrix operand order repaired and checked by reading, not compiled, not run on hardware
//
#include <hip/hip_runtime.h>

// ---------------------------------------------------------------------------
// VQ-VAE codebook quantization for MI455X (gfx1250, wave32, WMMA).
// Dominant op: [16384 x 256] @ [256 x 8192]^T with running argmin.
// Uses bf16 hi/lo split (3 WMMAs per chunk) for fp32-grade argmin accuracy
// on the bf16 matrix pipe: acc += zh*eh + zh*el + zl*eh  (fp32 accumulate).
// Two independent accumulator chains + explicit one-chunk B-fragment
// double buffering so WMMAs never drain loadcnt to zero.
// ---------------------------------------------------------------------------

#define NROWS   16384      // B*H*W = 16*32*32
#define KCODES  8192
#define CDIM    256
#define KSPLIT  2
#define BETA    0.25f

typedef __attribute__((ext_vector_type(16))) __bf16 v16bf;
typedef __attribute__((ext_vector_type(8)))  __bf16 bf16x8;
typedef __attribute__((ext_vector_type(8)))  float  v8f;

union ABfrag { v16bf v; bf16x8 h[2]; };

// ---------------------------------------------------------------------------
// Kernel 1: z [16][256][32][32] fp32 -> z_hi/z_lo bf16 in [n][c] row-major,
// n = b*1024 + h*32 + w.  LDS 32x32 tile transpose, coalesced both sides.
// grid = (32 hw-tiles, 8 c-tiles, 16 b), block = 256
// ---------------------------------------------------------------------------
__global__ __launch_bounds__(256)
void prep_z(const float* __restrict__ z, __bf16* __restrict__ zh,
            __bf16* __restrict__ zl) {
  __shared__ float tile[32][33];
  const int b   = blockIdx.z;
  const int c0  = blockIdx.y * 32;
  const int hw0 = blockIdx.x * 32;
  const int tx  = threadIdx.x & 31;
  const int ty  = threadIdx.x >> 5;            // 0..7
  const float* src = z + (size_t)b * (CDIM * 1024);
#pragma unroll
  for (int i = 0; i < 32; i += 8)
    tile[ty + i][tx] = src[(c0 + ty + i) * 1024 + hw0 + tx];
  __syncthreads();
#pragma unroll
  for (int i = 0; i < 32; i += 8) {
    const int hw = hw0 + ty + i;
    const int c  = c0 + tx;
    const float x  = tile[tx][ty + i];
    const __bf16 hi = (__bf16)x;
    const float  lo = x - (float)hi;
    const int o = (b * 1024 + hw) * CDIM + c;
    zh[o] = hi;
    zl[o] = (__bf16)lo;
  }
}

// ---------------------------------------------------------------------------
// Kernel 2: emb [8192][256] fp32 -> e_hi/e_lo bf16 + half_enorm[k] = 0.5*||e||^2
// one wave per codebook row; grid = 1024, block = 256 (8 waves)
// ---------------------------------------------------------------------------
__global__ __launch_bounds__(256)
void prep_e(const float* __restrict__ e, __bf16* __restrict__ eh,
            __bf16* __restrict__ el, float* __restrict__ he) {
  const int wave = threadIdx.x >> 5;
  const int lane = threadIdx.x & 31;
  const int k = blockIdx.x * 8 + wave;
  const float* row = e + k * CDIM;
  float s = 0.0f;
#pragma unroll
  for (int i = 0; i < 8; ++i) {
    const int c = i * 32 + lane;
    const float x = row[c];
    s += x * x;
    const __bf16 hi = (__bf16)x;
    eh[k * CDIM + c] = hi;
    el[k * CDIM + c] = (__bf16)(x - (float)hi);
  }
#pragma unroll
  for (int m = 16; m >= 1; m >>= 1) s += __shfl_xor(s, m, 32);
  if (lane == 0) he[k] = 0.5f * s;
}

// ---------------------------------------------------------------------------
// Kernel 3: GEMM + running argmax of t = z.e - 0.5||e||^2  (== argmin dist).
// grid = (128, KSPLIT); block = 256 (8 waves). Wave w owns rows
// [blk.x*128 + w*16, +16), scans its K half in 16-code steps.
// A frag (16x32 bf16): lane<16 holds row l15, K {c0..c0+7, c0+16..23};
//                      lane>=16 row l15, K {c0+8..15, c0+24..31}.
// B frag (32x16 bf16): lane<16 col l15 = code, K {c0..c0+7, c0+16..23};
//                      lane>=16 col l15, K {c0+8..15, c0+24..31}.
// C/D v8f: slot v, lane L -> row r0 + v + 8*(L>=16), col-code kb + (L&15).
// ---------------------------------------------------------------------------
__global__ __launch_bounds__(256)
void vq_gemm_argmin(const __bf16* __restrict__ zh, const __bf16* __restrict__ zl,
                    const __bf16* __restrict__ eh, const __bf16* __restrict__ el,
                    const float* __restrict__ he,
                    float* __restrict__ bestT, int* __restrict__ bestI) {
  const int wave = threadIdx.x >> 5;
  const int lane = threadIdx.x & 31;
  const int hfl  = lane >> 4;                  // 0 or 1
  const int l15  = lane & 15;
  const int r0   = blockIdx.x * 128 + wave * 16;
  const int ksp  = blockIdx.y;
  const int kbase = ksp * (KCODES / KSPLIT);

  // Hoist all A fragments (z rows are reused across the whole K loop).
  const int arow = r0 + l15;
  ABfrag ahi[8], alo[8];
#pragma unroll
  for (int cc = 0; cc < 8; ++cc) {
    const int c0 = cc * 32;
    const __bf16* p = zh + arow * CDIM + c0 + hfl * 8;
    ahi[cc].h[0] = *(const bf16x8*)p;
    ahi[cc].h[1] = *(const bf16x8*)(p + 16);
    const __bf16* q = zl + arow * CDIM + c0 + hfl * 8;
    alo[cc].h[0] = *(const bf16x8*)q;
    alo[cc].h[1] = *(const bf16x8*)(q + 16);
  }

  float tb[8];
  int   ib[8];
#pragma unroll
  for (int v = 0; v < 8; ++v) { tb[v] = -3.0e38f; ib[v] = 0; }

  for (int kt = 0; kt < KCODES / KSPLIT; kt += 16) {
    const int kb   = kbase + kt;
    const int krow = kb + l15;                 // this lane's code column
    const float h = he[krow];                  // hides under the WMMAs below
    const __bf16* pbase = eh + krow * CDIM + hfl * 8;
    const __bf16* qbase = el + krow * CDIM + hfl * 8;

    v8f acc0 = {};                             // chain 0: zh*eh + zl*eh
    v8f acc1 = {};                             // chain 1: zh*el

    // Explicit one-chunk software pipeline on the B fragments.
    ABfrag bhi, blo, bhiN, bloN;
    bhi.h[0] = *(const bf16x8*)(pbase);
    bhi.h[1] = *(const bf16x8*)(pbase + 16);
    blo.h[0] = *(const bf16x8*)(qbase);
    blo.h[1] = *(const bf16x8*)(qbase + 16);
#pragma unroll
    for (int cc = 0; cc < 8; ++cc) {
      if (cc < 7) {
        const int c1 = (cc + 1) * 32;
        bhiN.h[0] = *(const bf16x8*)(pbase + c1);
        bhiN.h[1] = *(const bf16x8*)(pbase + c1 + 16);
        bloN.h[0] = *(const bf16x8*)(qbase + c1);
        bloN.h[1] = *(const bf16x8*)(qbase + c1 + 16);
      }
      acc0 = __builtin_amdgcn_wmma_f32_16x16x32_bf16(false, ahi[cc].v, false, bhi.v,
                                                     (short)0, acc0, false, false);
      acc1 = __builtin_amdgcn_wmma_f32_16x16x32_bf16(false, ahi[cc].v, false, blo.v,
                                                     (short)0, acc1, false, false);
      acc0 = __builtin_amdgcn_wmma_f32_16x16x32_bf16(false, alo[cc].v, false, bhi.v,
                                                     (short)0, acc0, false, false);
      bhi = bhiN;                              // renamed away by full unroll
      blo = bloN;
    }
#pragma unroll
    for (int v = 0; v < 8; ++v) {
      const float t = (acc0[v] + acc1[v]) - h; // maximize z.e - 0.5||e||^2
      if (t > tb[v]) { tb[v] = t; ib[v] = krow; }
    }
  }

  // Reduce over the 16 code columns (masks < 16 keep the two half-groups
  // independent); tie-break toward the lower code index like jnp.argmin.
#pragma unroll
  for (int m = 8; m >= 1; m >>= 1) {
#pragma unroll
    for (int v = 0; v < 8; ++v) {
      const float t2 = __shfl_xor(tb[v], m, 32);
      const int   i2 = __shfl_xor(ib[v], m, 32);
      if (t2 > tb[v] || (t2 == tb[v] && i2 < ib[v])) { tb[v] = t2; ib[v] = i2; }
    }
  }
  if (l15 == 0) {
#pragma unroll
    for (int v = 0; v < 8; ++v) {
      const int row = r0 + v + hfl * 8;
      bestT[ksp * NROWS + row] = tb[v];
      bestI[ksp * NROWS + row] = ib[v];
    }
  }
}

// ---------------------------------------------------------------------------
// Kernel 4: merge the KSPLIT partial argmins, emit idx (int + float), zero loss.
// grid = 64, block = 256
// ---------------------------------------------------------------------------
__global__ __launch_bounds__(256)
void merge_idx(const float* __restrict__ bestT, const int* __restrict__ bestI,
               int* __restrict__ idx, float* __restrict__ out_idx_f,
               float* __restrict__ loss_slot) {
  const int n = blockIdx.x * 256 + threadIdx.x;
  const float t0 = bestT[n], t1 = bestT[NROWS + n];
  const int   i0 = bestI[n], i1 = bestI[NROWS + n];
  const int k = (t1 > t0) ? i1 : i0;           // tie -> split 0 (lower indices)
  idx[n] = k;
  out_idx_f[n] = (float)k;
  if (n == 0) *loss_slot = 0.0f;
}

// ---------------------------------------------------------------------------
// Kernel 5: z_q = emb[idx] written back in BCHW + loss accumulation.
// loss = (1+beta) * mean((z_q - z)^2).  grid = 16384, block = 256.
// ---------------------------------------------------------------------------
__global__ __launch_bounds__(256)
void quant_loss(const float* __restrict__ z, const float* __restrict__ e,
                const int* __restrict__ idx, float* __restrict__ zq,
                float* __restrict__ loss_slot) {
  __shared__ float red[8];
  const int o  = blockIdx.x * 256 + threadIdx.x;   // flat BCHW index
  const int b  = o >> 18;
  const int c  = (o >> 10) & 255;
  const int hw = o & 1023;
  const int n  = b * 1024 + hw;
  const int k  = idx[n];
  const float q = e[k * CDIM + c];
  const float d = q - z[o];
  zq[o] = q;
  float s = d * d;
#pragma unroll
  for (int m = 16; m >= 1; m >>= 1) s += __shfl_xor(s, m, 32);
  const int lane = threadIdx.x & 31, wave = threadIdx.x >> 5;
  if (lane == 0) red[wave] = s;
  __syncthreads();
  if (threadIdx.x == 0) {
    float t = 0.0f;
#pragma unroll
    for (int w = 0; w < 8; ++w) t += red[w];
    atomicAdd(loss_slot, t * ((1.0f + BETA) / (float)(NROWS * CDIM)));
  }
}

// ---------------------------------------------------------------------------
extern "C" void kernel_launch(void* const* d_in, const int* in_sizes, int n_in,
                              void* d_out, int out_size, void* d_ws, size_t ws_size,
                              hipStream_t stream) {
  const float* z   = (const float*)d_in[0];   // [16,256,32,32]
  const float* emb = (const float*)d_in[1];   // [8192,256]

  // Workspace layout (bytes)
  char* ws = (char*)d_ws;
  __bf16* zh = (__bf16*)(ws);                                   //  8 MB
  __bf16* zl = (__bf16*)(ws + (size_t)8  * 1024 * 1024);        //  8 MB
  __bf16* eh = (__bf16*)(ws + (size_t)16 * 1024 * 1024);        //  4 MB
  __bf16* el = (__bf16*)(ws + (size_t)20 * 1024 * 1024);        //  4 MB
  float*  he = (float*) (ws + (size_t)24 * 1024 * 1024);        // 32 KB
  float*  bT = (float*) (ws + (size_t)24 * 1024 * 1024 + (128 << 10));   // 128 KB
  int*    bI = (int*)   (ws + (size_t)24 * 1024 * 1024 + (256 << 10));   // 128 KB
  int*    ix = (int*)   (ws + (size_t)24 * 1024 * 1024 + (384 << 10));   //  64 KB

  float* out_zq   = (float*)d_out;                       // 4,194,304
  float* out_idx  = out_zq + (size_t)NROWS * CDIM;       // 16,384
  float* out_loss = out_idx + NROWS;                     // 1

  prep_z<<<dim3(32, 8, 16), 256, 0, stream>>>(z, zh, zl);
  prep_e<<<dim3(1024), 256, 0, stream>>>(emb, eh, el, he);
  vq_gemm_argmin<<<dim3(128, KSPLIT), 256, 0, stream>>>(zh, zl, eh, el, he, bT, bI);
  merge_idx<<<dim3(64), 256, 0, stream>>>(bT, bI, ix, out_idx, out_loss);
  quant_loss<<<dim3(16384), 256, 0, stream>>>(z, emb, ix, out_zq, out_loss);
}
